// ForceEGNN_11355893531208
// MI455X (gfx1250) — matrix operand order repaired and checked by reading, not compiled, not run on hardware
//
#include <hip/hip_runtime.h>
#include <hip/hip_bf16.h>

// ---------------------------------------------------------------------------
// ForceEGNN for MI455X (gfx1250).
// ~52 GFLOP of pairwise-MLP GEMM vs ~1MB of data -> pure compute bound.
// Dense 64x64 per-edge GEMMs run on v_wmma_f32_16x16x32_f16; activation
// re-layout goes through an m-major LDS tile so each WMMA A-fragment is
// exactly two ds_load_b128s. Rank-2 geometric term + silu on native VALU.
// ---------------------------------------------------------------------------

typedef __attribute__((ext_vector_type(16))) _Float16 v16h;
typedef __attribute__((ext_vector_type(8)))  float    v8f;
typedef __attribute__((ext_vector_type(4)))  unsigned u32x4;
typedef __attribute__((ext_vector_type(4)))  float    f32x4;

constexpr int BB = 2;
constexpr int NN = 512;
constexpr int HH = 64;
constexpr int LL = 3;

constexpr int WAVES_PER_BLOCK = 4;
constexpr int JCHUNK = 64;
constexpr int NJC    = NN / JCHUNK;   // 8
constexpr int ITILES = NN / 16;       // 32

__device__ __forceinline__ float fast_rcp(float x) {
#if defined(__has_builtin)
#if __has_builtin(__builtin_amdgcn_rcpf)
    return __builtin_amdgcn_rcpf(x);
#else
    return 1.0f / x;
#endif
#else
    return 1.0f / x;
#endif
}

__device__ __forceinline__ float fast_rsq(float x) {
#if defined(__has_builtin)
#if __has_builtin(__builtin_amdgcn_rsqf)
    return __builtin_amdgcn_rsqf(x);
#else
    return rsqrtf(x);
#endif
#else
    return rsqrtf(x);
#endif
}

__device__ __forceinline__ float silu_f(float x) {
    // x * sigmoid(x) with native exp + native rcp (no IEEE divide)
    return x * fast_rcp(1.0f + __expf(-x));
}

__device__ __forceinline__ v8f wmma_f16(v16h a, v16h b, v8f c) {
    // (neg_a, A, neg_b, B, c_mod, C, reuse_a, reuse_b)
    return __builtin_amdgcn_wmma_f32_16x16x32_f16(false, a, false, b,
                                                  (short)0, c, false, false);
}

// Same-wave LDS ordering: DS ops are in-order per wave on CDNA5; only need a
// compiler barrier (+ optional explicit dscnt wait).
__device__ __forceinline__ void lds_fence() {
#if defined(__has_builtin)
#if __has_builtin(__builtin_amdgcn_s_wait_dscnt)
    __builtin_amdgcn_s_wait_dscnt(0);
#endif
#endif
    asm volatile("" ::: "memory");
}

// B fragment (K x 16 slice of a (64,64) row-major f32 weight), per ISA layout:
// lane holds column n = nt*16 + (lane&15); k = ks*32 + 8*(lane>>4) + t (t<8), ks*32 + 16 + 8*(lane>>4) + t-8 (t>=8).
__device__ __forceinline__ v16h load_bfrag(const float* __restrict__ W,
                                           int n, int g, int ks) {
    v16h f;
#pragma unroll
    for (int t = 0; t < 16; ++t) {
        int k = ks * 32 + g * 8 + (t < 8 ? t : t + 8);
        f[t] = (_Float16)W[k * 64 + n];
    }
    return f;
}

// A fragment from an m-major LDS tile lds[m*64 + k] (16 rows x 64 channels,
// f16). Per the ISA 16-bit A layout, v16h elements 0..7 hold K = 8g+32ks+0..7
// and elements 8..15 hold K = 16+8g+32ks+0..7 for row m = lane&15 -> the
// whole fragment is two aligned ds_load_b128s.
union AFrag {
    struct { u32x4 lo, hi; } q;
    v16h h;
};
__device__ __forceinline__ v16h load_afrag128(const unsigned short* __restrict__ ldsw,
                                              int m, int g, int ks) {
    AFrag u;
    u.q.lo = *(const u32x4*)&ldsw[m * 64 + 8 * g + 32 * ks];
    u.q.hi = *(const u32x4*)&ldsw[m * 64 + 16 + 8 * g + 32 * ks];
    return u.h;
}

// ---------------------------------------------------------------------------
// init: h0 = silu(speed * h_init_w + b), speed_sq buffer, zero force accum.
// ---------------------------------------------------------------------------
__global__ __launch_bounds__(64) void k_init(const float* __restrict__ state,
                                             const float* __restrict__ hiw,
                                             const float* __restrict__ hib,
                                             float* __restrict__ h,
                                             float* __restrict__ sp,
                                             float* __restrict__ fr) {
    int row = blockIdx.x;
    int n = threadIdx.x;
    float vx = state[row * 4 + 2], vy = state[row * 4 + 3];
    float s2 = vx * vx + vy * vy;
    float spd = sqrtf(s2);
    h[row * HH + n] = silu_f(spd * hiw[n] + hib[n]);
    if (n == 0) sp[row] = s2;
    if (n < 2) fr[row * 2 + n] = 0.0f;
}

// ---------------------------------------------------------------------------
// per-layer: u[i] = h@W1[0:64] + b1 + sp_i*W1[129]; v[j] = h@W1[64:128] + sp_j*W1[130]
// Also zeroes m_agg for this layer.
// ---------------------------------------------------------------------------
__global__ __launch_bounds__(64) void k_uv(const float* __restrict__ h,
                                           const float* __restrict__ sp,
                                           const float* __restrict__ w1,
                                           const float* __restrict__ b1,
                                           float* __restrict__ u,
                                           float* __restrict__ v,
                                           float* __restrict__ mg) {
    __shared__ float hs[HH];
    int row = blockIdx.x, n = threadIdx.x;
    hs[n] = h[row * HH + n];
    __syncthreads();
    float s2 = sp[row];
    float au = b1[n] + s2 * w1[129 * 64 + n];
    float av = s2 * w1[130 * 64 + n];
#pragma unroll 8
    for (int k = 0; k < HH; ++k) {
        au += hs[k] * w1[k * 64 + n];
        av += hs[k] * w1[(64 + k) * 64 + n];
    }
    u[row * HH + n] = au;
    v[row * HH + n] = av;
    mg[row * HH + n] = 0.0f;
}

// ---------------------------------------------------------------------------
// Edge kernel: one wave = (b, 16-row i tile, 64-wide j chunk).
// Per j: rank-2 geo update (VALU, D-layout) -> t1 -> WMMA edge layer 2 (8)
//        -> m -> WMMA force layer 1 (8) -> dot fw2 -> force/normal accumulate.
// ---------------------------------------------------------------------------
__global__ __launch_bounds__(WAVES_PER_BLOCK * 32)
void k_edge(const float* __restrict__ state,
            const float* __restrict__ uu,
            const float* __restrict__ vv,
            const float* __restrict__ w1,   // (132,64) layer slice
            const float* __restrict__ w2,   // (64,64)
            const float* __restrict__ b2,   // (64)
            const float* __restrict__ fw1,  // (64,64)
            const float* __restrict__ fb1,  // (64)
            const float* __restrict__ fw2,  // (64)
            const float* __restrict__ fb2,  // (1)
            float* __restrict__ magg,       // (B*N,64) atomically accumulated
            float* __restrict__ force)      // (B*N,2)  atomically accumulated
{
    __shared__ alignas(16) unsigned short smem[WAVES_PER_BLOCK][16 * 64];

    const int lane = threadIdx.x & 31;
    const int wv   = threadIdx.x >> 5;
    const int wid  = blockIdx.x * WAVES_PER_BLOCK + wv;

    const int jc = wid % NJC;
    const int it = (wid / NJC) % ITILES;
    const int b  = wid / (NJC * ITILES);
    const int i0 = it * 16;
    const int j0 = jc * JCHUNK;

    unsigned short* ldsw = smem[wv];

    const int nb = lane & 15;   // n (column) base within 16, also A-row m
    const int g  = lane >> 4;   // lane half

    const f32x4* st4 = (const f32x4*)state;

    // --- per-lane i-side state for D-layout rows (m = r + 8g) ---
    float px_r[8], py_r[8], vx_r[8], vy_r[8];
#pragma unroll
    for (int r = 0; r < 8; ++r) {
        f32x4 s = st4[b * NN + i0 + r + 8 * g];
        px_r[r] = s.x; py_r[r] = s.y; vx_r[r] = s.z; vy_r[r] = s.w;
    }

    // --- preload u' in D-layout orientation ---
    float ud[4][8];
#pragma unroll
    for (int nt = 0; nt < 4; ++nt)
#pragma unroll
        for (int r = 0; r < 8; ++r)
            ud[nt][r] = uu[(size_t)(b * NN + i0 + r + 8 * g) * HH + nt * 16 + nb];

    // --- biases / fw2 / geo weight rows, broadcast per column ---
    float eb2c[4], fb1c[4], fw2r[4], w1c0[4], w1c3[4];
#pragma unroll
    for (int nt = 0; nt < 4; ++nt) {
        int n = nt * 16 + nb;
        eb2c[nt] = b2[n];
        fb1c[nt] = fb1[n];
        fw2r[nt] = fw2[n];
        w1c0[nt] = w1[128 * 64 + n];  // dist_sq row of edge_w1
        w1c3[nt] = w1[131 * 64 + n];  // approach row of edge_w1
    }
    const float fb2s = fb2[0];

    // --- weight B fragments (resident in VGPRs) ---
    v16h bw2[2][4], bfw1[2][4];
#pragma unroll
    for (int ks = 0; ks < 2; ++ks)
#pragma unroll
        for (int nt = 0; nt < 4; ++nt) {
            int n = nt * 16 + nb;
            bw2[ks][nt]  = load_bfrag(w2,  n, g, ks);
            bfw1[ks][nt] = load_bfrag(fw1, n, g, ks);
        }

    float macc[4][8];
#pragma unroll
    for (int nt = 0; nt < 4; ++nt)
#pragma unroll
        for (int r = 0; r < 8; ++r) macc[nt][r] = 0.0f;
    float fxa[8], fya[8];
#pragma unroll
    for (int r = 0; r < 8; ++r) { fxa[r] = 0.0f; fya[r] = 0.0f; }

    for (int j = j0; j < j0 + JCHUNK; ++j) {
        f32x4 sj = st4[b * NN + j];
        const float pxj = sj.x, pyj = sj.y, vxj = sj.z, vyj = sj.w;

        float vpr[4];
#pragma unroll
        for (int nt = 0; nt < 4; ++nt)
            vpr[nt] = vv[(size_t)(b * NN + j) * HH + nt * 16 + nb];

        // geometric scalars per D-layout row (rsqrt instead of divide);
        // rinv_r is reused by the force/normal update below.
        float dsq_r[8], apr_r[8], rinv_r[8];
#pragma unroll
        for (int r = 0; r < 8; ++r) {
            float dx = pxj - px_r[r], dy = pyj - py_r[r];
            float dsq = dx * dx + dy * dy;
            float rinv = fast_rsq(dsq + 1e-8f);
            dsq_r[r] = dsq;
            rinv_r[r] = rinv;
            apr_r[r] = ((vxj - vx_r[r]) * dx + (vyj - vy_r[r]) * dy) * rinv;
        }

        // ---- t1 = silu(u_i + v_j + dsq*W1[128] + appr*W1[131]) ----
        // computed directly in D-layout orientation, staged to LDS (m-major)
#pragma unroll
        for (int nt = 0; nt < 4; ++nt) {
            int n = nt * 16 + nb;
#pragma unroll
            for (int r = 0; r < 8; ++r) {
                float a = silu_f(ud[nt][r] + vpr[nt]
                                 + dsq_r[r] * w1c0[nt] + apr_r[r] * w1c3[nt]);
                ldsw[(r + 8 * g) * 64 + n] =
                    __builtin_bit_cast(unsigned short, (_Float16)a);
            }
        }
        lds_fence();

        v16h at0 = load_afrag128(ldsw, nb, g, 0);
        v16h at1 = load_afrag128(ldsw, nb, g, 1);

        // ---- GEMM2: m = silu(t1 @ W2 + b2): 8 WMMAs ----
        v8f d2[4];
#pragma unroll
        for (int nt = 0; nt < 4; ++nt) {
            v8f c;
#pragma unroll
            for (int r = 0; r < 8; ++r) c[r] = eb2c[nt];
            c = wmma_f16(at0, bw2[0][nt], c);
            c = wmma_f16(at1, bw2[1][nt], c);
            d2[nt] = c;
        }

        lds_fence();
        // silu + diagonal mask accumulate + restage to LDS
#pragma unroll
        for (int nt = 0; nt < 4; ++nt) {
            int n = nt * 16 + nb;
#pragma unroll
            for (int r = 0; r < 8; ++r) {
                float a = silu_f(d2[nt][r]);
                int m = r + 8 * g;
                macc[nt][r] += ((i0 + m) == j) ? 0.0f : a;
                ldsw[m * 64 + n] =
                    __builtin_bit_cast(unsigned short, (_Float16)a);
            }
        }
        lds_fence();

        v16h am0 = load_afrag128(ldsw, nb, g, 0);
        v16h am1 = load_afrag128(ldsw, nb, g, 1);

        // ---- GEMM3: gsil = silu(m @ fw1 + fb1): 8 WMMAs ----
        v8f d3[4];
#pragma unroll
        for (int nt = 0; nt < 4; ++nt) {
            v8f c;
#pragma unroll
            for (int r = 0; r < 8; ++r) c[r] = fb1c[nt];
            c = wmma_f16(am0, bfw1[0][nt], c);
            c = wmma_f16(am1, bfw1[1][nt], c);
            d3[nt] = c;
        }

        // ---- force scalar: dot with fw2, reduce over the 16 columns ----
        float pr[8];
#pragma unroll
        for (int r = 0; r < 8; ++r) pr[r] = 0.0f;
#pragma unroll
        for (int nt = 0; nt < 4; ++nt)
#pragma unroll
            for (int r = 0; r < 8; ++r)
                pr[r] += silu_f(d3[nt][r]) * fw2r[nt];
#pragma unroll
        for (int msk = 1; msk < 16; msk <<= 1)
#pragma unroll
            for (int r = 0; r < 8; ++r)
                pr[r] += __shfl_xor(pr[r], msk, 32);

#pragma unroll
        for (int r = 0; r < 8; ++r) {
            int ir = i0 + r + 8 * g;
            if (ir != j) {
                float fe = pr[r] + fb2s;
                float w = fe * rinv_r[r];      // fe / dist
                fxa[r] += w * (pxj - px_r[r]);
                fya[r] += w * (pyj - py_r[r]);
            }
        }
    }

    // commit m_agg (each (m,n) element owned by exactly one lane)
#pragma unroll
    for (int nt = 0; nt < 4; ++nt)
#pragma unroll
        for (int r = 0; r < 8; ++r)
            atomicAdd(&magg[(size_t)(b * NN + i0 + r + 8 * g) * HH + nt * 16 + nb],
                      macc[nt][r]);

    // force: values replicated across the 16 columns; one lane per half commits
    if (nb == 0) {
#pragma unroll
        for (int r = 0; r < 8; ++r) {
            int ir = i0 + r + 8 * g;
            atomicAdd(&force[(b * NN + ir) * 2 + 0], fxa[r]);
            atomicAdd(&force[(b * NN + ir) * 2 + 1], fya[r]);
        }
    }
}

// ---------------------------------------------------------------------------
// node update: h = lin(silu(lin([h, m_agg], nw1, nb1)), nw2, nb2)
// ---------------------------------------------------------------------------
__global__ __launch_bounds__(64) void k_node(float* __restrict__ h,
                                             const float* __restrict__ mg,
                                             const float* __restrict__ w1,
                                             const float* __restrict__ b1,
                                             const float* __restrict__ w2,
                                             const float* __restrict__ b2) {
    __shared__ float hs[HH], ms[HH], ts[HH];
    int row = blockIdx.x, n = threadIdx.x;
    hs[n] = h[row * HH + n];
    ms[n] = mg[row * HH + n];
    __syncthreads();
    float t = b1[n];
#pragma unroll 8
    for (int k = 0; k < HH; ++k)
        t += hs[k] * w1[k * 64 + n] + ms[k] * w1[(64 + k) * 64 + n];
    ts[n] = silu_f(t);
    __syncthreads();
    float o = b2[n];
#pragma unroll 8
    for (int k = 0; k < HH; ++k) o += ts[k] * w2[k * 64 + n];
    h[row * HH + n] = o;
}

// ---------------------------------------------------------------------------
// wall MLP + final output
// ---------------------------------------------------------------------------
__global__ __launch_bounds__(64) void k_wall(const float* __restrict__ state,
                                             const float* __restrict__ h,
                                             const float* __restrict__ fr,
                                             const float* __restrict__ w1,
                                             const float* __restrict__ b1,
                                             const float* __restrict__ w2,
                                             const float* __restrict__ b2,
                                             const float* __restrict__ w3,
                                             const float* __restrict__ b3,
                                             float* __restrict__ out) {
    __shared__ float xs[HH + 4], t1[HH], t2[HH], mag[4];
    int row = blockIdx.x, n = threadIdx.x;
    xs[n] = h[row * HH + n];
    if (n < 4) {
        float px = state[row * 4 + 0], py = state[row * 4 + 1];
        float wd = (n == 0) ? px : (n == 1) ? (1.0f - px)
                 : (n == 2) ? py : (1.0f - py);
        xs[HH + n] = wd;
    }
    __syncthreads();
    float a = b1[n];
#pragma unroll 4
    for (int k = 0; k < HH + 4; ++k) a += xs[k] * w1[k * 64 + n];
    t1[n] = silu_f(a);
    __syncthreads();
    float c = b2[n];
#pragma unroll 8
    for (int k = 0; k < HH; ++k) c += t1[k] * w2[k * 64 + n];
    t2[n] = silu_f(c);
    __syncthreads();
    if (n < 4) {
        float m = b3[n];
        for (int k = 0; k < HH; ++k) m += t2[k] * w3[k * 4 + n];
        mag[n] = m;
    }
    __syncthreads();
    if (n < 2) {
        float wf = (n == 0) ? (mag[0] - mag[1]) : (mag[2] - mag[3]);
        out[row * 2 + n] = fr[row * 2 + n] + wf;
    }
}

// ---------------------------------------------------------------------------
extern "C" void kernel_launch(void* const* d_in, const int* in_sizes, int n_in,
                              void* d_out, int out_size, void* d_ws, size_t ws_size,
                              hipStream_t stream) {
    (void)in_sizes; (void)n_in; (void)out_size; (void)ws_size;

    const float* state = (const float*)d_in[0];
    const float* hiw   = (const float*)d_in[1];
    const float* hib   = (const float*)d_in[2];
    const float* ew1   = (const float*)d_in[3];   // (L,132,64)
    const float* eb1   = (const float*)d_in[4];   // (L,64)
    const float* ew2   = (const float*)d_in[5];   // (L,64,64)
    const float* eb2   = (const float*)d_in[6];   // (L,64)
    const float* fw1   = (const float*)d_in[7];   // (L,64,64)
    const float* fb1   = (const float*)d_in[8];   // (L,64)
    const float* fw2   = (const float*)d_in[9];   // (L,64,1)
    const float* fb2   = (const float*)d_in[10];  // (L,1)
    const float* nw1   = (const float*)d_in[11];  // (L,128,64)
    const float* nb1   = (const float*)d_in[12];  // (L,64)
    const float* nw2   = (const float*)d_in[13];  // (L,64,64)
    const float* nb2   = (const float*)d_in[14];  // (L,64)
    const float* ww1   = (const float*)d_in[15];  // (68,64)
    const float* wb1   = (const float*)d_in[16];
    const float* ww2   = (const float*)d_in[17];  // (64,64)
    const float* wb2   = (const float*)d_in[18];
    const float* ww3   = (const float*)d_in[19];  // (64,4)
    const float* wb3   = (const float*)d_in[20];

    float* ws = (float*)d_ws;
    float* h  = ws;                       // B*N*H
    float* sp = h  + (size_t)BB * NN * HH;
    float* u  = sp + (size_t)BB * NN;
    float* v  = u  + (size_t)BB * NN * HH;
    float* mg = v  + (size_t)BB * NN * HH;
    float* fr = mg + (size_t)BB * NN * HH;

    const int rows = BB * NN;

    k_init<<<rows, 64, 0, stream>>>(state, hiw, hib, h, sp, fr);

    const int nwaves = BB * ITILES * NJC;                  // 512
    const int eblocks = nwaves / WAVES_PER_BLOCK;          // 128

    for (int l = 0; l < LL; ++l) {
        const float* w1l  = ew1 + (size_t)l * 132 * 64;
        k_uv<<<rows, 64, 0, stream>>>(h, sp, w1l, eb1 + l * 64, u, v, mg);
        k_edge<<<eblocks, WAVES_PER_BLOCK * 32, 0, stream>>>(
            state, u, v, w1l,
            ew2 + (size_t)l * 64 * 64, eb2 + l * 64,
            fw1 + (size_t)l * 64 * 64, fb1 + l * 64,
            fw2 + (size_t)l * 64, fb2 + l,
            mg, fr);
        k_node<<<rows, 64, 0, stream>>>(h, mg,
            nw1 + (size_t)l * 128 * 64, nb1 + l * 64,
            nw2 + (size_t)l * 64 * 64, nb2 + l * 64);
    }

    k_wall<<<rows, 64, 0, stream>>>(state, h, fr, ww1, wb1, ww2, wb2, ww3, wb3,
                                    (float*)d_out);
}
